// Decoder_35064113004947
// MI455X (gfx1250) — matrix operand order repaired and checked by reading, not compiled, not run on hardware
//
#include <hip/hip_runtime.h>
#include <hip/hip_bf16.h>

typedef __attribute__((ext_vector_type(16))) _Float16 v16h;
typedef __attribute__((ext_vector_type(8)))  _Float16 v8h;
typedef __attribute__((ext_vector_type(8)))  float    v8f;

#define NN    272      // graph nodes
#define OHS   16
#define CH    128
#define DIN   144      // 128 + OHS
#define KPAD  160      // DIN padded to multiple of 32 for WMMA K
#define V0C   34816    // NN * 128 vertices after layer0 upsample
#define V1C   69632    // NN * 256 vertices after layer1 upsample
#define E0C   (V0C * 8)
#define E1C   (V1C * 8)

__device__ __forceinline__ v16h cat8(v8h lo, v8h hi) {
  return __builtin_shufflevector(lo, hi, 0,1,2,3,4,5,6,7,8,9,10,11,12,13,14,15);
}

// ---------------------------------------------------------------------------
// Weight prep: conv weights [CH, DIN, 3] -> per-lane WMMA A-fragment order.
// Fragment index: ((mt*3 + k)*5 + kk)*512 + lane*16 + j
// A 16-bit layout: lane<16: M=lane, K(j<8)= (lane>>4)*8+j ; K(j>=8)=16+(lane>>4)*8+(j-8)
// ---------------------------------------------------------------------------
__global__ void prep_conv_w_kernel(const float* __restrict__ cw,
                                   _Float16* __restrict__ out) {
  int idx = blockIdx.x * blockDim.x + threadIdx.x;
  if (idx >= 8 * 15 * 512) return;
  int tile = idx >> 9, r = idx & 511;
  int lane = r >> 4, j = r & 15;
  int mt = tile / 15, rem = tile % 15;
  int k = rem / 5, kk = rem % 5;
  int ln15 = lane & 15, lhi = lane >> 4;
  int K = (j < 8) ? (lhi * 8 + j) : (16 + lhi * 8 + (j - 8));
  int i = kk * 32 + K;
  int ch = mt * 16 + ln15;
  float v = (i < DIN) ? cw[(ch * DIN + i) * 3 + k] : 0.f;
  out[idx] = (_Float16)v;
}

// mlp weights [288, CH]: build Wa = W_top - W_bot (for xi) and Wc = W_bot (for xj)
// as per-lane WMMA B-fragments. B layout: N = lane&15, K = (lane>>4)*16 + j.
// Fragment index: (nt*5 + kk)*512 + lane*16 + j
__global__ void prep_mlp_w_kernel(const float* __restrict__ mw,
                                  _Float16* __restrict__ wa,
                                  _Float16* __restrict__ wc) {
  int idx = blockIdx.x * blockDim.x + threadIdx.x;
  if (idx >= 2 * 40 * 512) return;
  int which = idx / (40 * 512);
  int r2 = idx % (40 * 512);
  int tile = r2 >> 9, r = r2 & 511;
  int lane = r >> 4, j = r & 15;
  int ch = (tile / 5) * 16 + (lane & 15);
  int K = (tile % 5) * 32 + (lane >> 4) * 16 + j;
  float v = 0.f;
  if (K < DIN) {
    float top = mw[K * CH + ch];
    float bot = mw[(DIN + K) * CH + ch];
    v = (which == 0) ? (top - bot) : bot;
  }
  (which == 0 ? wa : wc)[r2] = (_Float16)v;
}

// ---------------------------------------------------------------------------
// Conv1d(144->128,k=3,pad=1) + bias + ReLU + upsample x2 + concat(ohs) -> xv f16
// One block per node n. 256 threads = 8 waves; wave w owns ch-tile [16w,16w+16).
// Input xin: [NN, L, 128] row-major f32. xv: [NN*2L, KPAD] f16, cols 144..159 = 0.
// ---------------------------------------------------------------------------
__global__ __launch_bounds__(256) void conv_xv_kernel(
    const float* __restrict__ xin, const float* __restrict__ ohs,
    const float* __restrict__ cb, const _Float16* __restrict__ wfrag,
    _Float16* __restrict__ xv, int L) {
  extern __shared__ _Float16 sm[];  // (L+2) * KPAD, zero-padded rows 0 and L+1
  const int n = blockIdx.x;
  const int tid = threadIdx.x;
  const int L2 = 2 * L;

  const int tot = (L + 2) * KPAD;
  for (int idx = tid; idx < tot; idx += 256) {
    int r = idx / KPAD, c = idx % KPAD;
    float v = 0.f;
    if (r >= 1 && r <= L) {
      int t = r - 1;
      if (c < 128)       v = xin[((size_t)(n * L + t)) * 128 + c];
      else if (c < DIN)  v = ohs[n * OHS + (c - 128)];
    }
    sm[idx] = (_Float16)v;
  }
  // fill xv cols 128..159 (ohs + zero pad) for this node's 2L vertex rows
  for (int idx = tid; idx < L2 * 32; idx += 256) {
    int t2 = idx / 32, c = 128 + (idx % 32);
    float v = (c < DIN) ? ohs[n * OHS + (c - 128)] : 0.f;
    xv[(size_t)(n * L2 + t2) * KPAD + c] = (_Float16)v;
  }
  __syncthreads();

  const int wave = tid >> 5, lane = tid & 31;
  const int ln15 = lane & 15, lhi = lane >> 4;
  const int mbase = wave * 16;
  const int ntiles = L >> 4;

  for (int nt = 0; nt < ntiles; ++nt) {
    const int t = nt * 16 + ln15;  // time column owned by this lane (B N-dim)
    v8f acc = {};
#pragma unroll
    for (int k = 0; k < 3; ++k) {
      const _Float16* lrow = &sm[(size_t)(t + k) * KPAD];
#pragma unroll
      for (int kk = 0; kk < 5; ++kk) {
        v16h A = *(const v16h*)(wfrag + (((wave * 3 + k) * 5 + kk) << 9) + lane * 16);
        const _Float16* bp = lrow + kk * 32 + lhi * 8;
        v16h B = cat8(*(const v8h*)bp, *(const v8h*)(bp + 16));
        acc = __builtin_amdgcn_wmma_f32_16x16x32_f16(
            false, A, false, B, (short)0, acc, false, false);
      }
    }
    // epilogue: bias + relu, duplicate (nearest upsample) into xv rows 2t, 2t+1
#pragma unroll
    for (int g = 0; g < 8; ++g) {
      int ch = mbase + g + (lhi ? 8 : 0);
      float v = acc[g] + cb[ch];
      v = v > 0.f ? v : 0.f;
      _Float16 h = (_Float16)v;
      size_t vtx = (size_t)(n * L2 + 2 * t);
      xv[vtx * KPAD + ch] = h;
      xv[(vtx + 1) * KPAD + ch] = h;
    }
  }
}

// ---------------------------------------------------------------------------
// Per-vertex products: a = xv @ (Wtop - Wbot) + bias ; c = xv @ Wbot
// Grid: V/16 blocks, 8 waves/block; wave w owns ch-tile w. M = vertex, N = ch.
// ---------------------------------------------------------------------------
__global__ __launch_bounds__(256) void mlp_gemm_kernel(
    const _Float16* __restrict__ xv, const _Float16* __restrict__ wa,
    const _Float16* __restrict__ wc, const float* __restrict__ mb,
    float* __restrict__ aout, float* __restrict__ cout) {
  const int vb = blockIdx.x * 16;
  const int tid = threadIdx.x, wave = tid >> 5, lane = tid & 31;
  const int ln15 = lane & 15, lhi = lane >> 4;

  // A-fragments from xv: row = vb + ln15; two 8-half chunks per kk.
  v16h Af[5];
  const _Float16* xr = xv + (size_t)(vb + ln15) * KPAD + lhi * 16;
#pragma unroll
  for (int kk = 0; kk < 5; ++kk)
    Af[kk] = cat8(*(const v8h*)(xr + kk * 32), *(const v8h*)(xr + kk * 32 + 8));

  float bias = mb[wave * 16 + ln15];
  v8f acca, accc = {};
#pragma unroll
  for (int g = 0; g < 8; ++g) acca[g] = bias;

#pragma unroll
  for (int kk = 0; kk < 5; ++kk) {
    v16h Ba = *(const v16h*)(wa + ((wave * 5 + kk) << 9) + lane * 16);
    v16h Bc = *(const v16h*)(wc + ((wave * 5 + kk) << 9) + lane * 16);
    acca = __builtin_amdgcn_wmma_f32_16x16x32_f16(false, Af[kk], false, Ba,
                                                  (short)0, acca, false, false);
    accc = __builtin_amdgcn_wmma_f32_16x16x32_f16(false, Af[kk], false, Bc,
                                                  (short)0, accc, false, false);
  }
#pragma unroll
  for (int g = 0; g < 8; ++g) {
    int v = vb + g + (lhi ? 8 : 0);
    int ch = wave * 16 + ln15;
    aout[(size_t)v * CH + ch] = acca[g];
    cout[(size_t)v * CH + ch] = accc[g];
  }
}

// ------------------------------ CSR build ----------------------------------
__global__ void zero_int_kernel(int* p, int n) {
  for (int i = blockIdx.x * blockDim.x + threadIdx.x; i < n;
       i += gridDim.x * blockDim.x)
    p[i] = 0;
}
__global__ void hist_kernel(const int* __restrict__ dst, int E, int* counts) {
  for (int i = blockIdx.x * blockDim.x + threadIdx.x; i < E;
       i += gridDim.x * blockDim.x)
    atomicAdd(&counts[dst[i]], 1);
}
__global__ __launch_bounds__(1024) void scan_kernel(
    const int* __restrict__ counts, int V, int* __restrict__ offs,
    int* __restrict__ cursor) {
  __shared__ int smem[1024];
  int tid = threadIdx.x;
  int chunk = (V + 1023) / 1024;
  int s0 = tid * chunk;
  int s1 = s0 + chunk; if (s1 > V) s1 = V; if (s0 > V) s0 = V;
  int sum = 0;
  for (int i = s0; i < s1; ++i) sum += counts[i];
  smem[tid] = sum;
  __syncthreads();
  for (int off = 1; off < 1024; off <<= 1) {
    int v = (tid >= off) ? smem[tid - off] : 0;
    __syncthreads();
    smem[tid] += v;
    __syncthreads();
  }
  int run = smem[tid] - sum;  // exclusive base
  for (int i = s0; i < s1; ++i) {
    offs[i] = run; cursor[i] = run;
    run += counts[i];
  }
  if (tid == 1023) offs[V] = smem[1023];
}
__global__ void scatter_kernel(const int* __restrict__ src,
                               const int* __restrict__ dst, int E,
                               int* cursor, int* __restrict__ csr_src) {
  for (int i = blockIdx.x * blockDim.x + threadIdx.x; i < E;
       i += gridDim.x * blockDim.x) {
    int pos = atomicAdd(&cursor[dst[i]], 1);
    csr_src[pos] = src[i];
  }
}

// ---------------------------------------------------------------------------
// EdgeConv max-aggregation: one wave per dst vertex; lane owns 4 channels.
// agg[v] = max(0, max_j relu(a[v] + c[src_j]))  (relu>=0 so 0-init == ref)
// ---------------------------------------------------------------------------
__global__ __launch_bounds__(256) void edge_kernel(
    const float* __restrict__ aprod, const float* __restrict__ cprod,
    const int* __restrict__ offs, const int* __restrict__ csr_src,
    float* __restrict__ out, int V, int transposed) {
  int gw = (blockIdx.x * 256 + threadIdx.x) >> 5;
  int lane = threadIdx.x & 31;
  if (gw >= V) return;
  int s = offs[gw], e = offs[gw + 1];
  float4 a = ((const float4*)(aprod + (size_t)gw * CH))[lane];
  float4 acc = make_float4(0.f, 0.f, 0.f, 0.f);
  for (int j = s; j < e; ++j) {
    int sv = csr_src[j];
    float4 c = ((const float4*)(cprod + (size_t)sv * CH))[lane];
    acc.x = fmaxf(acc.x, fmaxf(a.x + c.x, 0.f));
    acc.y = fmaxf(acc.y, fmaxf(a.y + c.y, 0.f));
    acc.z = fmaxf(acc.z, fmaxf(a.z + c.z, 0.f));
    acc.w = fmaxf(acc.w, fmaxf(a.w + c.w, 0.f));
  }
  if (!transposed) {
    ((float4*)(out + (size_t)gw * CH))[lane] = acc;
  } else {  // final layer: v = n*256 + t  ->  out[t, n, ch]
    int n = gw >> 8, t = gw & 255;
    ((float4*)(out + ((size_t)t * NN + n) * CH))[lane] = acc;
  }
}

// ---------------------------------------------------------------------------
extern "C" void kernel_launch(void* const* d_in, const int* in_sizes, int n_in,
                              void* d_out, int out_size, void* d_ws,
                              size_t ws_size, hipStream_t stream) {
  (void)in_sizes; (void)n_in; (void)out_size; (void)ws_size;
  const float* data = (const float*)d_in[0];
  const float* ohs  = (const float*)d_in[1];   // [1,B,NP,OHS] == [NN, OHS] flat
  const int*   ei0  = (const int*)d_in[2];     // [2, E0]: src row0, dst row1
  const int*   ei1  = (const int*)d_in[3];
  const float* cw0  = (const float*)d_in[4];
  const float* cb0  = (const float*)d_in[5];
  const float* mw0  = (const float*)d_in[6];
  const float* mb0  = (const float*)d_in[7];
  const float* cw1  = (const float*)d_in[8];
  const float* cb1  = (const float*)d_in[9];
  const float* mw1  = (const float*)d_in[10];
  const float* mb1  = (const float*)d_in[11];
  float* outp = (float*)d_out;

  // workspace carve (256B aligned)
  char* w = (char*)d_ws;
  auto take = [&](size_t bytes) {
    char* p = w;
    w += (bytes + 255) & ~(size_t)255;
    return p;
  };
  _Float16* xv    = (_Float16*)take((size_t)V1C * KPAD * 2);
  float*    aprod = (float*)take((size_t)V1C * CH * 4);
  float*    cprod = (float*)take((size_t)V1C * CH * 4);
  float*    x1    = (float*)take((size_t)V0C * CH * 4);
  _Float16* cwf0  = (_Float16*)take(8 * 15 * 512 * 2);
  _Float16* cwf1  = (_Float16*)take(8 * 15 * 512 * 2);
  _Float16* wa0   = (_Float16*)take(40 * 512 * 2);
  _Float16* wc0   = (_Float16*)take(40 * 512 * 2);
  _Float16* wa1   = (_Float16*)take(40 * 512 * 2);
  _Float16* wc1   = (_Float16*)take(40 * 512 * 2);
  int*      counts = (int*)take((size_t)V1C * 4);
  int*      offs   = (int*)take((size_t)(V1C + 1) * 4);
  int*      cursor = (int*)take((size_t)V1C * 4);
  int*      csrsrc = (int*)take((size_t)E1C * 4);

  // ---- weight prep ----
  prep_conv_w_kernel<<<(8 * 15 * 512 + 255) / 256, 256, 0, stream>>>(cw0, cwf0);
  prep_conv_w_kernel<<<(8 * 15 * 512 + 255) / 256, 256, 0, stream>>>(cw1, cwf1);
  prep_mlp_w_kernel<<<(2 * 40 * 512 + 255) / 256, 256, 0, stream>>>(mw0, wa0, wc0);
  prep_mlp_w_kernel<<<(2 * 40 * 512 + 255) / 256, 256, 0, stream>>>(mw1, wa1, wc1);

  // ---- layer 0 (L = 64 -> L2 = 128, V0 vertices) ----
  conv_xv_kernel<<<NN, 256, (64 + 2) * KPAD * 2, stream>>>(data, ohs, cb0, cwf0,
                                                           xv, 64);
  mlp_gemm_kernel<<<V0C / 16, 256, 0, stream>>>(xv, wa0, wc0, mb0, aprod, cprod);
  zero_int_kernel<<<(V0C + 255) / 256, 256, 0, stream>>>(counts, V0C);
  hist_kernel<<<(E0C + 255) / 256, 256, 0, stream>>>(ei0 + E0C, E0C, counts);
  scan_kernel<<<1, 1024, 0, stream>>>(counts, V0C, offs, cursor);
  scatter_kernel<<<(E0C + 255) / 256, 256, 0, stream>>>(ei0, ei0 + E0C, E0C,
                                                        cursor, csrsrc);
  edge_kernel<<<V0C / 8, 256, 0, stream>>>(aprod, cprod, offs, csrsrc, x1, V0C, 0);

  // ---- layer 1 (L = 128 -> L2 = 256, V1 vertices) ----
  conv_xv_kernel<<<NN, 256, (128 + 2) * KPAD * 2, stream>>>(x1, ohs, cb1, cwf1,
                                                            xv, 128);
  mlp_gemm_kernel<<<V1C / 16, 256, 0, stream>>>(xv, wa1, wc1, mb1, aprod, cprod);
  zero_int_kernel<<<(V1C + 255) / 256, 256, 0, stream>>>(counts, V1C);
  hist_kernel<<<(E1C + 255) / 256, 256, 0, stream>>>(ei1 + E1C, E1C, counts);
  scan_kernel<<<1, 1024, 0, stream>>>(counts, V1C, offs, cursor);
  scatter_kernel<<<(E1C + 255) / 256, 256, 0, stream>>>(ei1, ei1 + E1C, E1C,
                                                        cursor, csrsrc);
  edge_kernel<<<V1C / 8, 256, 0, stream>>>(aprod, cprod, offs, csrsrc, outp, V1C, 1);
}
